// HetSGG_56977036149416
// MI455X (gfx1250) — matrix operand order repaired and checked by reading, not compiled, not run on hardware
//
#include <hip/hip_runtime.h>
#include <hip/hip_bf16.h>

// HetSGG on MI455X (gfx1250): relation-grouped bf16 WMMA GEMMs + segment softmax.
// Problem constants (from reference): C=256, R=9, B=4. N and E derived from in_sizes.
// 64-row tiles: B fragments (per-wave private column strips of W) are reused across
// 4 row-subtiles in registers -> 4x less L2 traffic on W than a 16-row tile.

#define CDIM 256
#define RNUM 9
#define BNUM 4
#define TM   64          // rows (edges) per block tile; bins are TM-aligned

typedef __bf16 bf16x16 __attribute__((ext_vector_type(16)));
typedef float  f32x8   __attribute__((ext_vector_type(8)));

__device__ __forceinline__ unsigned short f2bf(float f) {
  unsigned int u = __float_as_uint(f);
  u += 0x7FFFu + ((u >> 16) & 1u);          // round-to-nearest-even
  return (unsigned short)(u >> 16);
}

__device__ __forceinline__ void atomicMaxF(float* addr, float val) {
  // float max via sign-aware integer atomics (global_atomic_max_i32 / min_u32)
  if (val >= 0.0f) atomicMax((int*)addr, __float_as_int(val));
  else             atomicMin((unsigned int*)addr, (unsigned int)__float_as_int(val));
}

// ---------------------------------------------------------------- fills
__global__ void fill_f32_kernel(float* p, float v, size_t n) {
  size_t i = (size_t)blockIdx.x * blockDim.x + threadIdx.x;
  if (i < n) p[i] = v;
}
__global__ void fill_i32_kernel(int* p, int v, size_t n) {
  size_t i = (size_t)blockIdx.x * blockDim.x + threadIdx.x;
  if (i < n) p[i] = v;
}

// ------------------------------------------------- W[r] = sum_b att[r,b]*basis_b
// Stored transposed + bf16: Wt[r][n][k] = W_r[k][n], so each WMMA B-fragment lane
// reads 16 contiguous bf16 (32 bytes) along K.
__global__ void build_wt_kernel(const float* __restrict__ basis,
                                const float* __restrict__ att,
                                unsigned short* __restrict__ Wt) {
  size_t idx = (size_t)blockIdx.x * 256 + threadIdx.x;   // r*65536 + k*256 + n
  int r = (int)(idx >> 16);
  int k = (int)((idx >> 8) & 255);
  int n = (int)(idx & 255);
  float s = 0.0f;
#pragma unroll
  for (int b = 0; b < BNUM; ++b)
    s += att[r * BNUM + b] * basis[((size_t)b * CDIM + k) * CDIM + n];
  Wt[((size_t)r * CDIM + n) * CDIM + k] = f2bf(s);
}

// ------------------------------------------------- counting sort by relation
__global__ void hist_kernel(const int* __restrict__ tf, const int* __restrict__ ti,
                            int* cntF, int* cntI, int E) {
  int e = blockIdx.x * blockDim.x + threadIdx.x;
  if (e < E) { atomicAdd(&cntF[tf[e]], 1); atomicAdd(&cntI[ti[e]], 1); }
}

__global__ void scan_kernel(const int* cntF, const int* cntI, int* offF, int* offI) {
  if (blockIdx.x == 0 && threadIdx.x == 0) {
    int a = 0, b = 0;
    offF[0] = 0; offI[0] = 0;
    for (int r = 0; r < RNUM; ++r) {
      a += (cntF[r] + TM - 1) & ~(TM - 1); offF[r + 1] = a;  // TM-aligned bins ->
      b += (cntI[r] + TM - 1) & ~(TM - 1); offI[r + 1] = b;  // tiles never straddle a relation
    }
  }
}

__global__ void scatter_kernel(const int* __restrict__ tf, const int* __restrict__ ti,
                               const int* __restrict__ offF, const int* __restrict__ offI,
                               int* fillF, int* fillI, int* permF, int* permI, int E) {
  int e = blockIdx.x * blockDim.x + threadIdx.x;
  if (e < E) {
    int f = tf[e]; permF[offF[f] + atomicAdd(&fillF[f], 1)] = e;
    int i = ti[e]; permI[offI[i] + atomicAdd(&fillI[i], 1)] = e;
  }
}

// ------------------------------------------------- core WMMA GEMM tile kernel
// One block = TM=64 edges (one relation bin) x 256 output cols, K=256.
// 8 waves; wave w owns cols [32w, 32w+32): per K-step it loads its two B fragments
// once and reuses them across 4 row-subtiles (8 WMMAs per 4 global_load_b128).
// mode 0: out[e] = ef[e] + 0.5*m        (s2r)
// mode 1: out[e] += 0.5*m               (o2r)
// mode 2: score[e] = leaky(m . aw[r] + ab[r]); atomic max into smax[seg*R+r]
// mode 3: agg[seg[e]] += alpha[e]*m     (atomic scatter)
__global__ __launch_bounds__(256) void gemm_tile_kernel(
    const int* __restrict__ perm, const int* __restrict__ boff,
    const int* __restrict__ rowidx,                 // gather X row via this (nullptr = identity)
    const float* __restrict__ X,
    const unsigned short* __restrict__ Wt,          // [R][256][256] bf16 bits, Wt[n][k]
    int mode,
    const float* __restrict__ ef, float* __restrict__ out,
    const float* __restrict__ aw, const float* __restrict__ ab,
    float* __restrict__ score, float* __restrict__ smax,
    const int* __restrict__ segarr, const float* __restrict__ alpha) {

  __shared__ unsigned short As[TM * 264];   // TM rows bf16, stride 264 (16B aligned, bank-shifted)
  __shared__ float scoreLds[TM];
  __shared__ int   permLds[TM];
  __shared__ int   segLds[TM];
  __shared__ float alphaLds[TM];

  const int t    = threadIdx.x;
  const int row0 = blockIdx.x * TM;
  if (row0 >= boff[RNUM]) return;

  int r = 0;                                        // relation of this (TM-aligned) tile
#pragma unroll
  for (int i = 1; i < RNUM; ++i) r += (row0 >= boff[i]) ? 1 : 0;

  if (t < TM) {
    int e2 = perm[row0 + t];
    permLds[t]  = e2;
    segLds[t]   = (segarr != nullptr && e2 >= 0) ? segarr[e2] : 0;
    alphaLds[t] = (alpha  != nullptr && e2 >= 0) ? alpha[e2]  : 0.0f;
    scoreLds[t] = 0.0f;
  }

  // stage A: gather TM rows of 256 f32, convert to bf16 into LDS
#pragma unroll
  for (int it = 0; it < TM / 16; ++it) {
    const int row = it * 16 + (t >> 4);
    const int cs  = (t & 15) << 4;
    const int e   = perm[row0 + row];
    unsigned int w[8];
    if (e >= 0) {
      const int xr = (rowidx != nullptr) ? rowidx[e] : e;
      const float4* src = reinterpret_cast<const float4*>(X + (size_t)xr * CDIM + cs);
#pragma unroll
      for (int q = 0; q < 4; ++q) {
        float4 v = src[q];
        w[q * 2 + 0] = (unsigned)f2bf(v.x) | ((unsigned)f2bf(v.y) << 16);
        w[q * 2 + 1] = (unsigned)f2bf(v.z) | ((unsigned)f2bf(v.w) << 16);
      }
    } else {
#pragma unroll
      for (int q = 0; q < 8; ++q) w[q] = 0u;
    }
    uint4* dst = reinterpret_cast<uint4*>(&As[row * 264 + cs]);
    uint4 q0; q0.x = w[0]; q0.y = w[1]; q0.z = w[2]; q0.w = w[3];
    uint4 q1; q1.x = w[4]; q1.y = w[5]; q1.z = w[6]; q1.w = w[7];
    dst[0] = q0; dst[1] = q1;
  }
  __syncthreads();

  const int lane = t & 31;
  const int wave = t >> 5;
  const int g    = lane >> 4;    // half-wave (ISA fragment group)
  const int col  = lane & 15;
  const int n0   = wave << 5;

  f32x8 acc[TM / 16][2];
#pragma unroll
  for (int rt = 0; rt < TM / 16; ++rt) {
#pragma unroll
    for (int c = 0; c < 2; ++c) {
      f32x8 z = {0.f, 0.f, 0.f, 0.f, 0.f, 0.f, 0.f, 0.f};
      acc[rt][c] = z;
    }
  }
  const unsigned short* wb = Wt + (size_t)r * CDIM * CDIM;

#pragma unroll
  for (int kk = 0; kk < 8; ++kk) {                  // K = 8 x 32
    // B 32x16 bf16 fragments: col = lane&15, K = g*16 + 0..15 contiguous in Wt[n][k]
    union { bf16x16 v; uint4 u[2]; } b0f, b1f;
    const int kb = kk * 32 + g * 16;
    const uint4* wp0 = reinterpret_cast<const uint4*>(wb + (size_t)(n0 + col) * CDIM + kb);
    b0f.u[0] = wp0[0]; b0f.u[1] = wp0[1];
    const uint4* wp1 = reinterpret_cast<const uint4*>(wb + (size_t)(n0 + 16 + col) * CDIM + kb);
    b1f.u[0] = wp1[0]; b1f.u[1] = wp1[1];
#pragma unroll
    for (int rt = 0; rt < TM / 16; ++rt) {
      // A 16x32 bf16 fragment: row = lane&15; K = g*16 + 0..15 contiguous (matches B)
      union { bf16x16 v; uint4 u[2]; } af;
      const int abase = (rt * 16 + col) * 264 + kk * 32 + g * 16;
      af.u[0] = *reinterpret_cast<const uint4*>(&As[abase]);
      af.u[1] = *reinterpret_cast<const uint4*>(&As[abase + 8]);
      acc[rt][0] = __builtin_amdgcn_wmma_f32_16x16x32_bf16(false, af.v, false, b0f.v, (short)0, acc[rt][0], false, false);
      acc[rt][1] = __builtin_amdgcn_wmma_f32_16x16x32_bf16(false, af.v, false, b1f.v, (short)0, acc[rt][1], false, false);
    }
  }

  // C/D layout: lane holds rows rt*16 + j + 8*g (j=0..7), column = lane&15
  if (mode == 0) {
#pragma unroll
    for (int rt = 0; rt < TM / 16; ++rt) {
#pragma unroll
      for (int j = 0; j < 8; ++j) {
        const int rl = rt * 16 + j + g * 8;
        const int e  = permLds[rl];
        if (e >= 0) {
          const size_t base = (size_t)e * CDIM + n0 + col;
          out[base]      = ef[base]      + 0.5f * acc[rt][0][j];
          out[base + 16] = ef[base + 16] + 0.5f * acc[rt][1][j];
        }
      }
    }
  } else if (mode == 1) {
#pragma unroll
    for (int rt = 0; rt < TM / 16; ++rt) {
#pragma unroll
      for (int j = 0; j < 8; ++j) {
        const int rl = rt * 16 + j + g * 8;
        const int e  = permLds[rl];
        if (e >= 0) {
          const size_t base = (size_t)e * CDIM + n0 + col;
          out[base]      += 0.5f * acc[rt][0][j];  // unique rows: no atomics needed
          out[base + 16] += 0.5f * acc[rt][1][j];
        }
      }
    }
  } else if (mode == 2) {
    const float a0 = aw[r * CDIM + n0 + col];
    const float a1 = aw[r * CDIM + n0 + 16 + col];
#pragma unroll
    for (int rt = 0; rt < TM / 16; ++rt)
#pragma unroll
      for (int j = 0; j < 8; ++j)
        atomicAdd(&scoreLds[rt * 16 + j + g * 8],
                  acc[rt][0][j] * a0 + acc[rt][1][j] * a1);
    __syncthreads();
    if (t < TM) {
      const int e = permLds[t];
      if (e >= 0) {
        float s = scoreLds[t] + ab[r];
        s = (s > 0.0f) ? s : 0.01f * s;            // leaky_relu(0.01)
        score[e] = s;
        atomicMaxF(&smax[(size_t)segLds[t] * RNUM + r], s);
      }
    }
  } else { // mode 3
#pragma unroll
    for (int rt = 0; rt < TM / 16; ++rt) {
#pragma unroll
      for (int j = 0; j < 8; ++j) {
        const int rl = rt * 16 + j + g * 8;
        const int e  = permLds[rl];
        if (e >= 0) {
          const float al = alphaLds[rl];
          float* dst = out + (size_t)segLds[rl] * CDIM + n0 + col;
          atomicAdd(dst,      al * acc[rt][0][j]);
          atomicAdd(dst + 16, al * acc[rt][1][j]);
        }
      }
    }
  }
}

// ------------------------------------------------- softmax pieces + finalize
__global__ void expsum_kernel(const float* __restrict__ score, const float* __restrict__ smax,
                              const int* __restrict__ seg, const int* __restrict__ rt,
                              float* __restrict__ ex, float* __restrict__ den,
                              int* __restrict__ mask, int E) {
  int e = blockIdx.x * blockDim.x + threadIdx.x;
  if (e < E) {
    int s = seg[e], tt = rt[e];
    size_t gi = (size_t)s * RNUM + tt;
    float v = __expf(score[e] - smax[gi]);
    ex[e] = v;
    atomicAdd(&den[gi], v);
    atomicOr(&mask[s], 1 << tt);
  }
}

__global__ void alpha_kernel(float* __restrict__ ex, const float* __restrict__ den,
                             const int* __restrict__ seg, const int* __restrict__ rt, int E) {
  int e = blockIdx.x * blockDim.x + threadIdx.x;
  if (e < E) ex[e] = ex[e] / den[(size_t)seg[e] * RNUM + rt[e]];
}

__global__ void final_kernel(const float* __restrict__ nf,
                             const float* __restrict__ aggS, const float* __restrict__ aggO,
                             const int* __restrict__ maskS, const int* __restrict__ maskO,
                             float* __restrict__ outN, size_t total) {
  size_t i = (size_t)blockIdx.x * blockDim.x + threadIdx.x;
  if (i < total) {
    int n = (int)(i >> 8);
    float dS = (float)__popc((unsigned)maskS[n]); if (dS == 0.f) dS = 1.f;
    float dO = (float)__popc((unsigned)maskO[n]); if (dO == 0.f) dO = 1.f;
    outN[i] = nf[i] + 0.5f * (aggS[i] / dS + aggO[i] / dO);
  }
}

// ------------------------------------------------- launch
extern "C" void kernel_launch(void* const* d_in, const int* in_sizes, int n_in,
                              void* d_out, int out_size, void* d_ws, size_t ws_size,
                              hipStream_t stream) {
  const int*   edge_index = (const int*)d_in[0];
  const float* nf   = (const float*)d_in[1];
  const float* ef   = (const float*)d_in[2];
  const int*   et_f = (const int*)d_in[3];
  const int*   et_i = (const int*)d_in[4];
  // d_in[5] edge_norm: unused by the reference
  const float* s2r_basis = (const float*)d_in[6];
  const float* s2r_att   = (const float*)d_in[7];
  const float* o2r_basis = (const float*)d_in[8];
  const float* o2r_att   = (const float*)d_in[9];
  const float* r2s_basis = (const float*)d_in[10];
  const float* r2s_att   = (const float*)d_in[11];
  const float* r2s_aw    = (const float*)d_in[12];
  const float* r2s_ab    = (const float*)d_in[13];
  const float* r2o_basis = (const float*)d_in[14];
  const float* r2o_att   = (const float*)d_in[15];
  const float* r2o_aw    = (const float*)d_in[16];
  const float* r2o_ab    = (const float*)d_in[17];

  const int E = in_sizes[0] / 2;
  const int N = in_sizes[1] / CDIM;
  const int* src = edge_index;
  const int* dst = edge_index + E;

  float* node_emb = (float*)d_out;
  float* rel_emb  = (float*)d_out + (size_t)N * CDIM;

  const int tiles = (E + TM * RNUM + TM - 1) / TM;   // padded per-relation bins
  const int padE  = tiles * TM;
  const size_t NR = (size_t)N * RNUM;
  const size_t NC = (size_t)N * CDIM;

  // workspace carve (~45 MB total)
  size_t off = 0;
  char* base = (char*)d_ws;
  auto carve = [&](size_t bytes) -> void* {
    void* p = base + off; off += (bytes + 255) & ~(size_t)255; return p;
  };
  unsigned short* Wt_s2r = (unsigned short*)carve((size_t)RNUM * CDIM * CDIM * 2);
  unsigned short* Wt_o2r = (unsigned short*)carve((size_t)RNUM * CDIM * CDIM * 2);
  unsigned short* Wt_r2s = (unsigned short*)carve((size_t)RNUM * CDIM * CDIM * 2);
  unsigned short* Wt_r2o = (unsigned short*)carve((size_t)RNUM * CDIM * CDIM * 2);
  int* permF = (int*)carve((size_t)padE * 4);
  int* permI = (int*)carve((size_t)padE * 4);
  int* cntF  = (int*)carve(RNUM * 4);
  int* cntI  = (int*)carve(RNUM * 4);
  int* fillF = (int*)carve(RNUM * 4);
  int* fillI = (int*)carve(RNUM * 4);
  int* offF  = (int*)carve((RNUM + 1) * 4);
  int* offI  = (int*)carve((RNUM + 1) * 4);
  float* score = (float*)carve((size_t)E * 4);
  float* ex    = (float*)carve((size_t)E * 4);
  float* smaxS = (float*)carve(NR * 4);
  float* denS  = (float*)carve(NR * 4);
  float* smaxO = (float*)carve(NR * 4);
  float* denO  = (float*)carve(NR * 4);
  float* aggS  = (float*)carve(NC * 4);
  float* aggO  = (float*)carve(NC * 4);
  int* maskS = (int*)carve((size_t)N * 4);
  int* maskO = (int*)carve((size_t)N * 4);

  dim3 blk(256);
  auto g1 = [](size_t n) { return dim3((unsigned)((n + 255) / 256)); };

  // --- init (all accumulators re-zeroed every call; no cross-call state) ---
  fill_i32_kernel<<<g1(padE), blk, 0, stream>>>(permF, -1, (size_t)padE);
  fill_i32_kernel<<<g1(padE), blk, 0, stream>>>(permI, -1, (size_t)padE);
  fill_i32_kernel<<<1, blk, 0, stream>>>(cntF, 0, RNUM);
  fill_i32_kernel<<<1, blk, 0, stream>>>(cntI, 0, RNUM);
  fill_i32_kernel<<<1, blk, 0, stream>>>(fillF, 0, RNUM);
  fill_i32_kernel<<<1, blk, 0, stream>>>(fillI, 0, RNUM);
  fill_f32_kernel<<<g1(NR), blk, 0, stream>>>(denS, 0.0f, NR);
  fill_f32_kernel<<<g1(NR), blk, 0, stream>>>(denO, 0.0f, NR);
  fill_f32_kernel<<<g1(NR), blk, 0, stream>>>(smaxS, -1e30f, NR);
  fill_f32_kernel<<<g1(NR), blk, 0, stream>>>(smaxO, -1e30f, NR);
  fill_f32_kernel<<<g1(NC), blk, 0, stream>>>(aggS, 0.0f, NC);
  fill_f32_kernel<<<g1(NC), blk, 0, stream>>>(aggO, 0.0f, NC);
  fill_i32_kernel<<<g1(N), blk, 0, stream>>>(maskS, 0, (size_t)N);
  fill_i32_kernel<<<g1(N), blk, 0, stream>>>(maskO, 0, (size_t)N);

  // --- relation weights (bf16, transposed) ---
  dim3 wgrid(RNUM * CDIM);
  build_wt_kernel<<<wgrid, blk, 0, stream>>>(s2r_basis, s2r_att, Wt_s2r);
  build_wt_kernel<<<wgrid, blk, 0, stream>>>(o2r_basis, o2r_att, Wt_o2r);
  build_wt_kernel<<<wgrid, blk, 0, stream>>>(r2s_basis, r2s_att, Wt_r2s);
  build_wt_kernel<<<wgrid, blk, 0, stream>>>(r2o_basis, r2o_att, Wt_r2o);

  // --- counting sort by relation (fwd + inv orderings) ---
  hist_kernel<<<g1(E), blk, 0, stream>>>(et_f, et_i, cntF, cntI, E);
  scan_kernel<<<1, 1, 0, stream>>>(cntF, cntI, offF, offI);
  scatter_kernel<<<g1(E), blk, 0, stream>>>(et_f, et_i, offF, offI, fillF, fillI, permF, permI, E);

  dim3 ggrid(tiles);
  // --- rel_emb = ef + 0.5*(s2r + o2r) ---
  gemm_tile_kernel<<<ggrid, blk, 0, stream>>>(permF, offF, src, nf, Wt_s2r, 0,
      ef, rel_emb, nullptr, nullptr, nullptr, nullptr, nullptr, nullptr);
  gemm_tile_kernel<<<ggrid, blk, 0, stream>>>(permI, offI, dst, nf, Wt_o2r, 1,
      nullptr, rel_emb, nullptr, nullptr, nullptr, nullptr, nullptr, nullptr);

  // --- branch rel2sub: types=inv, seg=src ---
  gemm_tile_kernel<<<ggrid, blk, 0, stream>>>(permI, offI, nullptr, rel_emb, Wt_r2s, 2,
      nullptr, nullptr, r2s_aw, r2s_ab, score, smaxS, src, nullptr);
  expsum_kernel<<<g1(E), blk, 0, stream>>>(score, smaxS, src, et_i, ex, denS, maskS, E);
  alpha_kernel<<<g1(E), blk, 0, stream>>>(ex, denS, src, et_i, E);
  gemm_tile_kernel<<<ggrid, blk, 0, stream>>>(permI, offI, nullptr, rel_emb, Wt_r2s, 3,
      nullptr, aggS, nullptr, nullptr, nullptr, nullptr, src, ex);

  // --- branch rel2obj: types=fwd, seg=dst ---
  gemm_tile_kernel<<<ggrid, blk, 0, stream>>>(permF, offF, nullptr, rel_emb, Wt_r2o, 2,
      nullptr, nullptr, r2o_aw, r2o_ab, score, smaxO, dst, nullptr);
  expsum_kernel<<<g1(E), blk, 0, stream>>>(score, smaxO, dst, et_f, ex, denO, maskO, E);
  alpha_kernel<<<g1(E), blk, 0, stream>>>(ex, denO, dst, et_f, E);
  gemm_tile_kernel<<<ggrid, blk, 0, stream>>>(permF, offF, nullptr, rel_emb, Wt_r2o, 3,
      nullptr, aggO, nullptr, nullptr, nullptr, nullptr, dst, ex);

  // --- node_emb = nf + 0.5*(aggS/divS + aggO/divO) ---
  final_kernel<<<g1(NC), blk, 0, stream>>>(nf, aggS, aggO, maskS, maskO, node_emb, NC);
}
